// AttentiveFPModel_11733850653136
// MI455X (gfx1250) — compile-verified, matrix operand order checked by reading, not run on hardware
//
#include <hip/hip_runtime.h>
#include <hip/hip_bf16.h>
#include <math.h>

typedef __attribute__((ext_vector_type(16))) _Float16 v16h;
typedef __attribute__((ext_vector_type(8)))  _Float16 v8h;
typedef __attribute__((ext_vector_type(8)))  float    v8f;

#define NNODE 100000
#define NEDGE 1600000
#define HDIM  64

__device__ __forceinline__ float lrelu(float v){ return v > 0.f ? v : 0.01f*v; }
__device__ __forceinline__ float eluf(float v){ return v > 0.f ? v : (expf(v)-1.f); }
__device__ __forceinline__ float sigm(float v){ return 1.f/(1.f+expf(-v)); }

__device__ __forceinline__ v8f wmma16(v16h a, v16h b, v8f c) {
  return __builtin_amdgcn_wmma_f32_16x16x32_f16(false, a, false, b, (short)0, c,
                                                false, false);
}

// A fragment (16x32 f16) from a contiguous 64-half row, per documented layout:
// lanes 0-15 hold K={kt*32+0..7, kt*32+16..23}; lanes 16-31 K offset by +8.
__device__ __forceinline__ v16h afrag_f16(const _Float16* row, int kt, int lane) {
  int kh = (lane & 16) ? 8 : 0;
  v8h lo = *(const v8h*)(row + kt*32 + kh);
  v8h hi = *(const v8h*)(row + kt*32 + 16 + kh);
  v16h r;
  #pragma unroll
  for (int i = 0; i < 8; ++i) { r[i] = lo[i]; r[i+8] = hi[i]; }
  return r;
}

// A fragment built from f32 row with elu(v + bias) applied on the fly.
__device__ __forceinline__ v16h afrag_f32_elu(const float* row, const float* bias,
                                              int kt, int lane) {
  int kh = (lane & 16) ? 8 : 0;
  int b0 = kt*32 + kh, b1 = kt*32 + 16 + kh;
  v16h r;
  #pragma unroll
  for (int i = 0; i < 8; ++i) {
    r[i]   = (_Float16)eluf(row[b0+i] + bias[b0+i]);
    r[i+8] = (_Float16)eluf(row[b1+i] + bias[b1+i]);
  }
  return r;
}

__device__ __forceinline__ v16h bfrag(const _Float16* buf, int tile, int lane) {
  return *(const v16h*)(buf + (size_t)tile*512 + lane*16);
}

__device__ __forceinline__ void atomicMaxF(float* addr, float val) {
  if (val >= 0.f) atomicMax((int*)addr, __float_as_int(val));
  else            atomicMin((unsigned int*)addr, __float_as_uint(val));
}

// ---------------- prep kernels ----------------

__global__ void fill_f32(float* p, float v, int n) {
  int i = blockIdx.x*blockDim.x + threadIdx.x;
  if (i < n) p[i] = v;
}

// Pack B fragments for y = x @ W^T  (B[k][n] = W[n][k]).  W is [n_out, ld].
__global__ void pack_bfrags(const float* __restrict__ W, _Float16* __restrict__ out,
                            int ktiles, int ntiles, int ld) {
  int tid = blockIdx.x*blockDim.x + threadIdx.x;
  int total = ktiles*ntiles*512;
  if (tid >= total) return;
  int e    = tid & 15;
  int lane = (tid >> 4) & 31;
  int tile = tid >> 9;
  int kt = tile / ntiles, nt = tile % ntiles;
  int n  = nt*16 + (lane & 15);
  int kh = (lane >> 4) * 8;
  int K  = kt*32 + ((e < 8) ? (e + kh) : (e + 8 + kh));
  out[tid] = (_Float16)W[(size_t)n*ld + K];
}

// x = lrelu(node_attr @ w1^T + b1)   (IN_DIM==1 -> outer product)
__global__ void prep_x(const float* __restrict__ na, const float* __restrict__ w1,
                       const float* __restrict__ b1, float* __restrict__ x32,
                       _Float16* __restrict__ x16) {
  int i = blockIdx.x*blockDim.x + threadIdx.x;
  if (i >= NNODE*HDIM) return;
  int n = i >> 6, c = i & 63;
  float v = lrelu(na[n]*w1[c] + b1[c]);
  x32[i] = v; x16[i] = (_Float16)v;
}

// ---------------- edge pass 1: h_e, a, segment-max ----------------

__global__ void edge_pass1(const _Float16* __restrict__ x16, const float* __restrict__ x32,
                           const float* __restrict__ edge_attr,
                           const int* __restrict__ src, const int* __restrict__ dst,
                           const _Float16* __restrict__ fG1, const float* __restrict__ wg1,
                           const float* __restrict__ att_l, const float* __restrict__ att_r,
                           float* __restrict__ a_out, float* __restrict__ amax) {
  int lane = threadIdx.x & 31;
  int wave = (blockIdx.x*blockDim.x + threadIdx.x) >> 5;
  int e0 = wave << 4;
  if (e0 >= NEDGE) return;
  int q = lane & 15;
  const _Float16* arow = x16 + (size_t)src[e0 + q]*HDIM;
  v16h a0 = afrag_f16(arow, 0, lane);
  v16h a1 = afrag_f16(arow, 1, lane);
  float ea[8];
  #pragma unroll
  for (int r = 0; r < 8; ++r) ea[r] = edge_attr[e0 + ((lane < 16) ? r : r + 8)];
  float s[8];
  #pragma unroll
  for (int r = 0; r < 8; ++r) s[r] = 0.f;
  #pragma unroll
  for (int nt = 0; nt < 4; ++nt) {
    v8f c = {0.f,0.f,0.f,0.f,0.f,0.f,0.f,0.f};
    c = wmma16(a0, bfrag(fG1, 0*4 + nt, lane), c);
    c = wmma16(a1, bfrag(fG1, 1*4 + nt, lane), c);
    int n = nt*16 + q;
    float wcol = wg1[n*65 + 64];
    float al   = att_l[n];
    #pragma unroll
    for (int r = 0; r < 8; ++r) {
      float he = lrelu(c[r] + ea[r]*wcol);
      s[r] += he*al;
    }
  }
  #pragma unroll
  for (int r = 0; r < 8; ++r)
    #pragma unroll
    for (int m = 1; m < 16; m <<= 1) s[r] += __shfl_xor(s[r], m, 32);
  if (q < 8) {
    float sv = 0.f;
    #pragma unroll
    for (int r = 0; r < 8; ++r) if (r == q) sv = s[r];
    int m = q + ((lane < 16) ? 0 : 8);
    int e = e0 + m;
    int d = dst[e];
    const float* xd = x32 + (size_t)d*HDIM;
    float hr = 0.f;
    for (int k = 0; k < HDIM; ++k) hr += xd[k]*att_r[k];
    float av = lrelu(sv + hr);
    a_out[e] = av;
    atomicMaxF(amax + d, av);
  }
}

// ---------------- edge pass 2: exp + denom ----------------

__global__ void edge_pass2(const float* __restrict__ a_in, const int* __restrict__ dst,
                           const float* __restrict__ amax,
                           float* __restrict__ e_out, float* __restrict__ denom) {
  int e = blockIdx.x*blockDim.x + threadIdx.x;
  if (e >= NEDGE) return;
  int d = dst[e];
  float ev = expf(a_in[e] - amax[d]);
  e_out[e] = ev;
  atomicAdd(denom + d, ev);
}

// ---------------- edge pass 3: (h_e @ wg2^T) * alpha scatter-add ----------------

#define P3_WAVES 4
__global__ void edge_pass3(const _Float16* __restrict__ x16, const float* __restrict__ edge_attr,
                           const int* __restrict__ src, const int* __restrict__ dst,
                           const _Float16* __restrict__ fG1, const _Float16* __restrict__ fG2,
                           const float* __restrict__ wg1, const float* __restrict__ ealpha,
                           const float* __restrict__ denom, float* __restrict__ hout) {
  __shared__ __align__(16) _Float16 tile[P3_WAVES][16][80];
  int lane = threadIdx.x & 31;
  int w    = threadIdx.x >> 5;
  int wave = (blockIdx.x*blockDim.x + threadIdx.x) >> 5;
  int e0 = wave << 4;
  bool active = e0 < NEDGE;
  int q = lane & 15;
  if (active) {
    const _Float16* arow = x16 + (size_t)src[e0 + q]*HDIM;
    v16h a0 = afrag_f16(arow, 0, lane);
    v16h a1 = afrag_f16(arow, 1, lane);
    float ea[8];
    #pragma unroll
    for (int r = 0; r < 8; ++r) ea[r] = edge_attr[e0 + ((lane < 16) ? r : r + 8)];
    #pragma unroll
    for (int nt = 0; nt < 4; ++nt) {
      v8f c = {0.f,0.f,0.f,0.f,0.f,0.f,0.f,0.f};
      c = wmma16(a0, bfrag(fG1, 0*4 + nt, lane), c);
      c = wmma16(a1, bfrag(fG1, 1*4 + nt, lane), c);
      int n = nt*16 + q;
      float wcol = wg1[n*65 + 64];
      #pragma unroll
      for (int r = 0; r < 8; ++r) {
        int m = (lane < 16) ? r : r + 8;
        tile[w][m][n] = (_Float16)lrelu(c[r] + ea[r]*wcol);
      }
    }
  }
  __syncthreads();
  if (active) {
    const _Float16* hrow = &tile[w][q][0];
    v16h a0 = afrag_f16(hrow, 0, lane);
    v16h a1 = afrag_f16(hrow, 1, lane);
    int de[8]; float al[8];
    #pragma unroll
    for (int r = 0; r < 8; ++r) {
      int e = e0 + ((lane < 16) ? r : r + 8);
      de[r] = dst[e];
      al[r] = ealpha[e] / denom[de[r]];
    }
    #pragma unroll
    for (int nt = 0; nt < 4; ++nt) {
      v8f c = {0.f,0.f,0.f,0.f,0.f,0.f,0.f,0.f};
      c = wmma16(a0, bfrag(fG2, 0*4 + nt, lane), c);
      c = wmma16(a1, bfrag(fG2, 1*4 + nt, lane), c);
      int n = nt*16 + q;
      #pragma unroll
      for (int r = 0; r < 8; ++r)
        atomicAdd(hout + (size_t)de[r]*HDIM + n, c[r]*al[r]);
    }
  }
}

// ---------------- node GRU1 (16 nodes per wave, 48 WMMAs) ----------------

__global__ void node_gru1(const float* __restrict__ h, const float* __restrict__ bg,
                          const _Float16* __restrict__ x16, const float* __restrict__ x32,
                          const _Float16* __restrict__ fIH, const _Float16* __restrict__ fHH,
                          const float* __restrict__ bih, const float* __restrict__ bhh,
                          float* __restrict__ x2, _Float16* __restrict__ x2h) {
  int lane = threadIdx.x & 31;
  int wave = (blockIdx.x*blockDim.x + threadIdx.x) >> 5;
  int n0 = wave << 4;
  if (n0 >= NNODE) return;
  int q = lane & 15;
  const float* hrow = h + (size_t)(n0 + q)*HDIM;
  v16h ap0 = afrag_f32_elu(hrow, bg, 0, lane);
  v16h ap1 = afrag_f32_elu(hrow, bg, 1, lane);
  const _Float16* xrow = x16 + (size_t)(n0 + q)*HDIM;
  v16h ax0 = afrag_f16(xrow, 0, lane);
  v16h ax1 = afrag_f16(xrow, 1, lane);
  #pragma unroll
  for (int nt = 0; nt < 4; ++nt) {
    v8f ir = {0.f,0.f,0.f,0.f,0.f,0.f,0.f,0.f};
    v8f iz = ir, inn = ir, hr = ir, hz = ir, hn = ir;
    ir  = wmma16(ap0, bfrag(fIH, 0*12 + nt,     lane), ir);
    ir  = wmma16(ap1, bfrag(fIH, 1*12 + nt,     lane), ir);
    iz  = wmma16(ap0, bfrag(fIH, 0*12 + nt + 4, lane), iz);
    iz  = wmma16(ap1, bfrag(fIH, 1*12 + nt + 4, lane), iz);
    inn = wmma16(ap0, bfrag(fIH, 0*12 + nt + 8, lane), inn);
    inn = wmma16(ap1, bfrag(fIH, 1*12 + nt + 8, lane), inn);
    hr  = wmma16(ax0, bfrag(fHH, 0*12 + nt,     lane), hr);
    hr  = wmma16(ax1, bfrag(fHH, 1*12 + nt,     lane), hr);
    hz  = wmma16(ax0, bfrag(fHH, 0*12 + nt + 4, lane), hz);
    hz  = wmma16(ax1, bfrag(fHH, 1*12 + nt + 4, lane), hz);
    hn  = wmma16(ax0, bfrag(fHH, 0*12 + nt + 8, lane), hn);
    hn  = wmma16(ax1, bfrag(fHH, 1*12 + nt + 8, lane), hn);
    int n = nt*16 + q;
    #pragma unroll
    for (int r = 0; r < 8; ++r) {
      int m = (lane < 16) ? r : r + 8;
      size_t idx = (size_t)(n0 + m)*HDIM + n;
      float rg = sigm(ir[r] + bih[n]       + hr[r] + bhh[n]);
      float zg = sigm(iz[r] + bih[64 + n]  + hz[r] + bhh[64 + n]);
      float ng = tanhf(inn[r] + bih[128 + n] + rg*(hn[r] + bhh[128 + n]));
      float ho = x32[idx];
      float o  = fmaxf((1.f - zg)*ng + zg*ho, 0.f);
      x2[idx]  = o;
      x2h[idx] = (_Float16)o;
    }
  }
}

// ---------------- column reductions (readout / weighted sum) ----------------

__global__ void colsum(const float* __restrict__ x, const float* __restrict__ scale,
                       float* __restrict__ out64, int rows) {
  __shared__ float red[256];
  int tid = blockIdx.x*blockDim.x + threadIdx.x;
  int c = threadIdx.x & 63;
  int rowStart  = tid >> 6;
  int rowStride = (gridDim.x*blockDim.x) >> 6;
  float p = 0.f;
  for (int r = rowStart; r < rows; r += rowStride) {
    float v = x[(size_t)r*HDIM + c];
    if (scale) v *= scale[r];
    p += v;
  }
  red[threadIdx.x] = p;
  __syncthreads();
  if (threadIdx.x < 64) {
    float s = red[threadIdx.x] + red[threadIdx.x+64] + red[threadIdx.x+128] + red[threadIdx.x+192];
    atomicAdd(out64 + threadIdx.x, s);
  }
}

// out0 = relu(sum); xd = out0 @ wm^T; c0 = xd . att_dst
__global__ void mid_kernel(const float* __restrict__ sum64, const float* __restrict__ wm,
                           const float* __restrict__ att_dst,
                           float* __restrict__ out0, float* __restrict__ c0) {
  __shared__ float o[64], xd[64];
  int c = threadIdx.x;
  float v = fmaxf(sum64[c], 0.f);
  o[c] = v; out0[c] = v;
  __syncthreads();
  float s = 0.f;
  for (int k = 0; k < 64; ++k) s += wm[c*64 + k]*o[k];
  xd[c] = s*att_dst[c];
  __syncthreads();
  if (c == 0) { float t = 0.f; for (int k = 0; k < 64; ++k) t += xd[k]; *c0 = t; }
}

// xs = x2 @ wm^T ; a2 = lrelu(xs . att_src + c0) ; global max
__global__ void node_xs(const _Float16* __restrict__ x2h, const _Float16* __restrict__ fWM,
                        const float* __restrict__ att_src, const float* __restrict__ c0,
                        float* __restrict__ xs, float* __restrict__ a2,
                        float* __restrict__ gmax) {
  int lane = threadIdx.x & 31;
  int wave = (blockIdx.x*blockDim.x + threadIdx.x) >> 5;
  int n0 = wave << 4;
  if (n0 >= NNODE) return;
  int q = lane & 15;
  const _Float16* xr = x2h + (size_t)(n0 + q)*HDIM;
  v16h a0 = afrag_f16(xr, 0, lane);
  v16h a1 = afrag_f16(xr, 1, lane);
  float s[8];
  #pragma unroll
  for (int r = 0; r < 8; ++r) s[r] = 0.f;
  #pragma unroll
  for (int nt = 0; nt < 4; ++nt) {
    v8f c = {0.f,0.f,0.f,0.f,0.f,0.f,0.f,0.f};
    c = wmma16(a0, bfrag(fWM, 0*4 + nt, lane), c);
    c = wmma16(a1, bfrag(fWM, 1*4 + nt, lane), c);
    int n = nt*16 + q;
    float as = att_src[n];
    #pragma unroll
    for (int r = 0; r < 8; ++r) {
      int m = (lane < 16) ? r : r + 8;
      xs[(size_t)(n0 + m)*HDIM + n] = c[r];
      s[r] += c[r]*as;
    }
  }
  #pragma unroll
  for (int r = 0; r < 8; ++r)
    #pragma unroll
    for (int m = 1; m < 16; m <<= 1) s[r] += __shfl_xor(s[r], m, 32);
  float av = -3.0e38f;
  if (q < 8) {
    float sv = 0.f;
    #pragma unroll
    for (int r = 0; r < 8; ++r) if (r == q) sv = s[r];
    int m = q + ((lane < 16) ? 0 : 8);
    float v = lrelu(sv + c0[0]);
    a2[n0 + m] = v;
    av = v;
  }
  #pragma unroll
  for (int m = 1; m < 32; m <<= 1) av = fmaxf(av, __shfl_xor(av, m, 32));
  if (lane == 0) atomicMaxF(gmax, av);
}

__global__ void node_exp(const float* __restrict__ a2, const float* __restrict__ gmax,
                         float* __restrict__ e2, float* __restrict__ gden) {
  int n = blockIdx.x*blockDim.x + threadIdx.x;
  float v = 0.f;
  if (n < NNODE) { v = expf(a2[n] - gmax[0]); e2[n] = v; }
  #pragma unroll
  for (int m = 1; m < 32; m <<= 1) v += __shfl_xor(v, m, 32);
  if ((threadIdx.x & 31) == 0) atomicAdd(gden, v);
}

// ---------------- final tiny stage: GRU2 + linear ----------------

__global__ void final_kernel(const float* __restrict__ h2sum, const float* __restrict__ gden,
                             const float* __restrict__ bm, const float* __restrict__ out0,
                             const float* __restrict__ wih, const float* __restrict__ whh,
                             const float* __restrict__ bih, const float* __restrict__ bhh,
                             const float* __restrict__ w2, const float* __restrict__ b2,
                             float* __restrict__ dout) {
  __shared__ float hp[64], hh[64], o[64];
  int c = threadIdx.x;
  float h2 = h2sum[c]/gden[0] + bm[c];
  hp[c] = eluf(h2);
  hh[c] = out0[c];
  __syncthreads();
  float ir=0.f, iz=0.f, inn=0.f, hr=0.f, hz=0.f, hn=0.f;
  for (int k = 0; k < 64; ++k) {
    float xp = hp[k], xh = hh[k];
    ir  += wih[(c)*64 + k]*xp;
    iz  += wih[(64 + c)*64 + k]*xp;
    inn += wih[(128 + c)*64 + k]*xp;
    hr  += whh[(c)*64 + k]*xh;
    hz  += whh[(64 + c)*64 + k]*xh;
    hn  += whh[(128 + c)*64 + k]*xh;
  }
  float rg = sigm(ir + bih[c] + hr + bhh[c]);
  float zg = sigm(iz + bih[64 + c] + hz + bhh[64 + c]);
  float ng = tanhf(inn + bih[128 + c] + rg*(hn + bhh[128 + c]));
  float on = fmaxf((1.f - zg)*ng + zg*hh[c], 0.f);
  o[c] = on;
  __syncthreads();
  float f = 0.f;
  for (int k = 0; k < 64; ++k) f += w2[c*64 + k]*o[k];
  dout[c] = f + b2[c];
}

// ---------------- launch ----------------

extern "C" void kernel_launch(void* const* d_in, const int* in_sizes, int n_in,
                              void* d_out, int out_size, void* d_ws, size_t ws_size,
                              hipStream_t stream) {
  const float* node_attr = (const float*)d_in[0];
  const float* edge_attr = (const float*)d_in[1];
  const int*   eidx      = (const int*)d_in[2];
  const float* w1   = (const float*)d_in[3];
  const float* b1   = (const float*)d_in[4];
  const float* wg1  = (const float*)d_in[5];
  const float* attl = (const float*)d_in[6];
  const float* attr_ = (const float*)d_in[7];
  const float* wg2  = (const float*)d_in[8];
  const float* bg   = (const float*)d_in[9];
  const float* g1wih = (const float*)d_in[10];
  const float* g1whh = (const float*)d_in[11];
  const float* g1bih = (const float*)d_in[12];
  const float* g1bhh = (const float*)d_in[13];
  const float* wm    = (const float*)d_in[14];
  const float* attsrc = (const float*)d_in[15];
  const float* attdst = (const float*)d_in[16];
  const float* bm     = (const float*)d_in[17];
  const float* g2wih = (const float*)d_in[18];
  const float* g2whh = (const float*)d_in[19];
  const float* g2bih = (const float*)d_in[20];
  const float* g2bhh = (const float*)d_in[21];
  const float* w2   = (const float*)d_in[22];
  const float* b2   = (const float*)d_in[23];
  const int* srcI = eidx;
  const int* dstI = eidx + NEDGE;

  char* wsp = (char*)d_ws;
  auto alloc = [&](size_t bytes) -> void* {
    void* p = (void*)wsp;
    wsp += (bytes + 255) & ~(size_t)255;
    return p;
  };
  float*     x32  = (float*)alloc((size_t)NNODE*HDIM*4);
  _Float16*  x16  = (_Float16*)alloc((size_t)NNODE*HDIM*2);
  float*     x2   = (float*)alloc((size_t)NNODE*HDIM*4);
  _Float16*  x2h  = (_Float16*)alloc((size_t)NNODE*HDIM*2);
  float*     hbuf = (float*)alloc((size_t)NNODE*HDIM*4);
  float*     xs   = (float*)alloc((size_t)NNODE*HDIM*4);
  float*     aE   = (float*)alloc((size_t)NEDGE*4);
  float*     eE   = (float*)alloc((size_t)NEDGE*4);
  float*     amax = (float*)alloc((size_t)NNODE*4);
  float*     den  = (float*)alloc((size_t)NNODE*4);
  float*     a2   = (float*)alloc((size_t)NNODE*4);
  float*     e2   = (float*)alloc((size_t)NNODE*4);
  _Float16*  fG1  = (_Float16*)alloc(8*512*2);
  _Float16*  fG2  = (_Float16*)alloc(8*512*2);
  _Float16*  fIH  = (_Float16*)alloc(24*512*2);
  _Float16*  fHH  = (_Float16*)alloc(24*512*2);
  _Float16*  fWM  = (_Float16*)alloc(8*512*2);
  float*     sum64 = (float*)alloc(64*4);
  float*     out0  = (float*)alloc(64*4);
  float*     c0    = (float*)alloc(4);
  float*     gmax  = (float*)alloc(4);
  float*     gden  = (float*)alloc(4);
  float*     h2sum = (float*)alloc(64*4);

  const float NEGINF = -__builtin_inff();
  // init
  fill_f32<<<(NNODE+255)/256, 256, 0, stream>>>(amax, NEGINF, NNODE);
  fill_f32<<<(NNODE+255)/256, 256, 0, stream>>>(den, 0.f, NNODE);
  fill_f32<<<(NNODE*HDIM+255)/256, 256, 0, stream>>>(hbuf, 0.f, NNODE*HDIM);
  fill_f32<<<1, 256, 0, stream>>>(sum64, 0.f, 64);
  fill_f32<<<1, 256, 0, stream>>>(h2sum, 0.f, 64);
  fill_f32<<<1, 32, 0, stream>>>(gmax, NEGINF, 1);
  fill_f32<<<1, 32, 0, stream>>>(gden, 0.f, 1);

  // pack weight fragments
  pack_bfrags<<<(8*512+255)/256, 256, 0, stream>>>(wg1, fG1, 2, 4, 65);
  pack_bfrags<<<(8*512+255)/256, 256, 0, stream>>>(wg2, fG2, 2, 4, 64);
  pack_bfrags<<<(24*512+255)/256, 256, 0, stream>>>(g1wih, fIH, 2, 12, 64);
  pack_bfrags<<<(24*512+255)/256, 256, 0, stream>>>(g1whh, fHH, 2, 12, 64);
  pack_bfrags<<<(8*512+255)/256, 256, 0, stream>>>(wm, fWM, 2, 4, 64);

  // x = lrelu(lin1(node_attr))
  prep_x<<<(NNODE*HDIM+255)/256, 256, 0, stream>>>(node_attr, w1, b1, x32, x16);

  // GATEConv
  {
    int waves = NEDGE/16;                      // 100000
    edge_pass1<<<waves/8, 256, 0, stream>>>(x16, x32, edge_attr, srcI, dstI,
                                            fG1, wg1, attl, attr_, aE, amax);
    edge_pass2<<<(NEDGE+255)/256, 256, 0, stream>>>(aE, dstI, amax, eE, den);
    edge_pass3<<<waves/P3_WAVES, P3_WAVES*32, 0, stream>>>(x16, edge_attr, srcI, dstI,
                                                           fG1, fG2, wg1, eE, den, hbuf);
  }

  // GRU1 + relu
  {
    int waves = NNODE/16;                      // 6250
    node_gru1<<<(waves+7)/8, 256, 0, stream>>>(hbuf, bg, x16, x32, fIH, fHH,
                                               g1bih, g1bhh, x2, x2h);
  }

  // readout sum -> out0, c0
  colsum<<<256, 256, 0, stream>>>(x2, (const float*)nullptr, sum64, NNODE);
  mid_kernel<<<1, 64, 0, stream>>>(sum64, wm, attdst, out0, c0);

  // GATConv over single segment
  {
    int waves = NNODE/16;
    node_xs<<<(waves+7)/8, 256, 0, stream>>>(x2h, fWM, attsrc, c0, xs, a2, gmax);
    node_exp<<<(NNODE+255)/256, 256, 0, stream>>>(a2, gmax, e2, gden);
    colsum<<<256, 256, 0, stream>>>(xs, e2, h2sum, NNODE);
  }

  // GRU2 + final linear
  final_kernel<<<1, 64, 0, stream>>>(h2sum, gden, bm, out0, g2wih, g2whh,
                                     g2bih, g2bhh, w2, b2, (float*)d_out);
}
